// SimpleSAGEConv_7490422964616
// MI455X (gfx1250) — matrix operand order repaired and checked by reading, not compiled, not run on hardware
//
#include <hip/hip_runtime.h>
#include <hip/hip_bf16.h>

#define N_NODES 100000
#define N_EDGES 1600000
#define NODE_DIM 128
#define EDGE_DIM 32
#define OUT_DIM 128
#define K_DIM 320            // NODE_DIM + 2*EDGE_DIM + NODE_DIM (agg 192 | x 128)
#define AGG_DIM 192
#define N_TILES (N_NODES / 16)   // 6250, exact

typedef __attribute__((ext_vector_type(16))) __bf16 v16bf;
typedef __attribute__((ext_vector_type(8)))  float  v8f;

union BFrag {
    v16bf v;
    __bf16 e[16];
    uint4  q[2];
};

// ---------------------------------------------------------------------------
// Kernel 1: zero the agg workspace (N_NODES * 192 floats), vectorized.
// ---------------------------------------------------------------------------
__global__ void zero_kernel(uint4* __restrict__ p, unsigned n4) {
    unsigned i = blockIdx.x * blockDim.x + threadIdx.x;
    unsigned stride = gridDim.x * blockDim.x;
    uint4 z = {0u, 0u, 0u, 0u};
    for (; i < n4; i += stride) p[i] = z;
}

// ---------------------------------------------------------------------------
// Kernel 2: build transposed bf16 fused weight table.
//   WT[n*320 + k] = bf16( k<192 ? 0.5*Wm[k][n] : Wr[k-192][n] )
// ---------------------------------------------------------------------------
__global__ void wprep_kernel(const float* __restrict__ Wm,
                             const float* __restrict__ Wr,
                             unsigned short* __restrict__ WT) {
    int idx = blockIdx.x * blockDim.x + threadIdx.x;
    if (idx >= K_DIM * OUT_DIM) return;
    int n = idx & (OUT_DIM - 1);
    int k = idx >> 7;
    float v = (k < AGG_DIM) ? 0.5f * Wm[k * OUT_DIM + n]
                            : Wr[(k - AGG_DIM) * OUT_DIM + n];
    WT[(size_t)n * K_DIM + k] = __builtin_bit_cast(unsigned short, (__bf16)v);
}

// ---------------------------------------------------------------------------
// Kernel 3: scatter-add the per-edge 192-wide message into agg[N,192] with
// hardware f32 atomics (agg fits in the 192MB L2; atomics resolve there).
// blockDim = 192 (6 waves): thread k owns message element k for 16 edges.
// ---------------------------------------------------------------------------
__global__ void scatter_kernel(const float* __restrict__ x,
                               const int*   __restrict__ ei,
                               const float* __restrict__ ea,
                               const float* __restrict__ et,
                               float* __restrict__ agg) {
    const int k = threadIdx.x;               // 0..191
    const int ebase = blockIdx.x * 16;
#pragma unroll 4
    for (int j = 0; j < 16; ++j) {
        const int e = ebase + j;
        const int r = ei[e];                 // edge_index[0][e] (dst)
        const int c = ei[N_EDGES + e];       // edge_index[1][e] (src)
        float v;
        if (k < NODE_DIM)
            v = x[(size_t)c * NODE_DIM + k];
        else if (k < NODE_DIM + EDGE_DIM)
            v = ea[(size_t)e * EDGE_DIM + (k - NODE_DIM)];
        else
            v = et[(size_t)e * EDGE_DIM + (k - NODE_DIM - EDGE_DIM)];
        unsafeAtomicAdd(&agg[(size_t)r * AGG_DIM + k], v);   // global_atomic_add_f32
    }
}

// ---------------------------------------------------------------------------
// Kernel 4: fused GEMM via v_wmma_f32_16x16x32_bf16.
//   out[m][n] = sum_k A[m][k] * Wcat[k][n] + (0.5*bm[n] + br[n])
//   A[m][k]   = k<192 ? agg[m][k] : x[m][k-192]
// One wave per 16-row tile; 10 K-tiles x 8 N-tiles of WMMA; f32 accumulate.
// Fragment layouts, both in k = 16g+s order (A: row=L&15, K run k0+16g+[0..15];
// B from transposed table: col=L&15, contiguous 16-elem K run; C: row=i+8g).
// ---------------------------------------------------------------------------
__device__ inline void load_run8(const float* __restrict__ agg,
                                 const float* __restrict__ x,
                                 int m, int k, float4& lo, float4& hi) {
    if (k < AGG_DIM) {
        const float4* p = (const float4*)(agg + (size_t)m * AGG_DIM + k);
        lo = p[0]; hi = p[1];
    } else {
        const float4* p = (const float4*)(x + (size_t)m * NODE_DIM + (k - AGG_DIM));
        lo = p[0]; hi = p[1];
    }
}

__global__ void gemm_kernel(const float* __restrict__ agg,
                            const float* __restrict__ x,
                            const unsigned short* __restrict__ WT,
                            const float* __restrict__ bm,
                            const float* __restrict__ br,
                            float* __restrict__ out) {
    const int lane = threadIdx.x & 31;
    const int wave = threadIdx.x >> 5;
    const int tile = blockIdx.x * 8 + wave;
    if (tile >= N_TILES) return;

    const int row0 = tile * 16;
    const int g  = lane >> 4;        // K-half / row-half selector
    const int ln = lane & 15;
    const int m  = row0 + ln;        // A row owned by this lane

    v8f zero8 = {0.f, 0.f, 0.f, 0.f, 0.f, 0.f, 0.f, 0.f};
    v8f acc[8];
#pragma unroll
    for (int t = 0; t < 8; ++t) acc[t] = zero8;

    for (int kt = 0; kt < K_DIM / 32; ++kt) {
        const int k0 = kt * 32;
        // A fragment: runs K = k0+16g+[0..7] and k0+16g+8+[0..7]
        float4 r0lo, r0hi, r1lo, r1hi;
        load_run8(agg, x, m, k0 + 16 * g,     r0lo, r0hi);
        load_run8(agg, x, m, k0 + 16 * g + 8, r1lo, r1hi);
        BFrag a;
        a.e[0] = (__bf16)r0lo.x;  a.e[1] = (__bf16)r0lo.y;
        a.e[2] = (__bf16)r0lo.z;  a.e[3] = (__bf16)r0lo.w;
        a.e[4] = (__bf16)r0hi.x;  a.e[5] = (__bf16)r0hi.y;
        a.e[6] = (__bf16)r0hi.z;  a.e[7] = (__bf16)r0hi.w;
        a.e[8]  = (__bf16)r1lo.x; a.e[9]  = (__bf16)r1lo.y;
        a.e[10] = (__bf16)r1lo.z; a.e[11] = (__bf16)r1lo.w;
        a.e[12] = (__bf16)r1hi.x; a.e[13] = (__bf16)r1hi.y;
        a.e[14] = (__bf16)r1hi.z; a.e[15] = (__bf16)r1hi.w;

#pragma unroll
        for (int nt = 0; nt < 8; ++nt) {
            // B fragment: column n = nt*16 + ln, K = k0 + 16g + [0..15] contiguous
            const uint4* bp = (const uint4*)(WT + (size_t)(nt * 16 + ln) * K_DIM
                                             + k0 + 16 * g);
            BFrag b;
            b.q[0] = bp[0];
            b.q[1] = bp[1];
            acc[nt] = __builtin_amdgcn_wmma_f32_16x16x32_bf16(
                false, a.v, false, b.v, (short)0, acc[nt], false, false);
        }
    }

    // Store with fused bias: out = acc + 0.5*bm + br
#pragma unroll
    for (int nt = 0; nt < 8; ++nt) {
        const int col = nt * 16 + ln;
        const float bias = 0.5f * bm[col] + br[col];
#pragma unroll
        for (int i = 0; i < 8; ++i) {
            const int r = row0 + i + 8 * g;      // C/D layout: M = i + 8g
            out[(size_t)r * OUT_DIM + col] = acc[nt][i] + bias;
        }
    }
}

// ---------------------------------------------------------------------------
extern "C" void kernel_launch(void* const* d_in, const int* in_sizes, int n_in,
                              void* d_out, int out_size, void* d_ws, size_t ws_size,
                              hipStream_t stream) {
    const float* x  = (const float*)d_in[0];
    const int*   ei = (const int*)  d_in[1];   // [2, E] int32
    const float* ea = (const float*)d_in[2];
    const float* et = (const float*)d_in[3];
    const float* Wm = (const float*)d_in[4];
    const float* bm = (const float*)d_in[5];
    const float* Wr = (const float*)d_in[6];
    const float* br = (const float*)d_in[7];
    float* out = (float*)d_out;

    // Workspace layout: agg fp32 [N,192] then WT bf16 [128,320]
    float* agg = (float*)d_ws;
    const size_t agg_bytes = (size_t)N_NODES * AGG_DIM * sizeof(float); // 76.8 MB, 16B aligned
    unsigned short* WT = (unsigned short*)((char*)d_ws + agg_bytes);

    const unsigned n4 = (unsigned)(N_NODES * AGG_DIM / 4);
    zero_kernel<<<2048, 256, 0, stream>>>((uint4*)agg, n4);

    wprep_kernel<<<(K_DIM * OUT_DIM + 255) / 256, 256, 0, stream>>>(Wm, Wr, WT);

    scatter_kernel<<<N_EDGES / 16, 192, 0, stream>>>(x, ei, ea, et, agg);

    gemm_kernel<<<(N_TILES + 7) / 8, 256, 0, stream>>>(agg, x, WT, bm, br, out);
}
